// DynamicConv2d_4638564680066
// MI455X (gfx1250) — matrix operand order repaired and checked by reading, not compiled, not run on hardware
//
#include <hip/hip_runtime.h>
#include <cstdint>
#include <cstddef>

typedef __attribute__((ext_vector_type(16))) __bf16 v16bf;
typedef __attribute__((ext_vector_type(8)))  float  v8f;
typedef __attribute__((ext_vector_type(4)))  unsigned int u32x4;
typedef __attribute__((ext_vector_type(4)))  int v4i;

#define B_     32
#define CIN_   256
#define COUT_  256
#define HW_    64
#define KEXP_  4
#define NTAP   9

// ---- CDNA5 async global->LDS path (guarded; fallback = register copy) ----
#if defined(__has_builtin)
#if __has_builtin(__builtin_amdgcn_global_load_async_to_lds_b128) && \
    __has_builtin(__builtin_amdgcn_s_wait_asynccnt)
#define USE_ASYNC_LDS 1
#endif
#endif
#ifndef USE_ASYNC_LDS
#define USE_ASYNC_LDS 0
#endif

#if USE_ASYNC_LDS
// pointer-to-AS1-v4i and pointer-to-AS3-v4i, per the builtin's signature
typedef __attribute__((address_space(1))) v4i GlobalV4;
typedef __attribute__((address_space(3))) v4i LdsV4;
#endif

union FragAB { v16bf v; u32x4 u[2]; };

// ---------------- Phase 1: attention (pool -> MLP -> softmax) ----------------
__global__ void attn_kernel(const float* __restrict__ x,
                            const float* __restrict__ fc1w, const float* __restrict__ fc1b,
                            const float* __restrict__ fc2w, const float* __restrict__ fc2b,
                            float* __restrict__ att) {
  __shared__ float pooled[CIN_];
  __shared__ float a1[KEXP_];
  int b = blockIdx.x, c = threadIdx.x;
  const float4* px = (const float4*)(x + (size_t)(b * CIN_ + c) * (HW_ * HW_));
  float s = 0.f;
  for (int i = 0; i < (HW_ * HW_) / 4; ++i) {
    float4 q = px[i];
    s += q.x + q.y + q.z + q.w;
  }
  pooled[c] = s * (1.0f / (HW_ * HW_));
  __syncthreads();
  if (c < KEXP_) {
    float acc = fc1b[c];
    for (int j = 0; j < CIN_; ++j) acc += fc1w[c * CIN_ + j] * pooled[j];
    a1[c] = acc > 0.f ? acc : 0.f;
  }
  __syncthreads();
  if (c == 0) {
    float a2[KEXP_];
    float m = -1e30f;
    for (int k = 0; k < KEXP_; ++k) {
      float acc = fc2b[k];
      for (int j = 0; j < KEXP_; ++j) acc += fc2w[k * KEXP_ + j] * a1[j];
      a2[k] = acc;
      m = fmaxf(m, acc);
    }
    float den = 0.f;
    for (int k = 0; k < KEXP_; ++k) { a2[k] = __expf(a2[k] - m); den += a2[k]; }
    for (int k = 0; k < KEXP_; ++k) att[b * KEXP_ + k] = a2[k] / den;
  }
}

// ---------------- Phase 2a: aggregated bias ----------------
__global__ void aggb_kernel(const float* __restrict__ att, const float* __restrict__ bias,
                            float* __restrict__ aggB) {
  int b = blockIdx.x, co = threadIdx.x;
  float s = 0.f;
  for (int k = 0; k < KEXP_; ++k) s += att[b * KEXP_ + k] * bias[k * COUT_ + co];
  aggB[b * COUT_ + co] = s;
}

// ---------------- Phase 2b: aggregated weights, bf16 hi/lo split ----------------
// weight layout: [K][COUT][CIN][3][3], output layout: [b][tap][COUT][CIN] (hi & lo)
__global__ void aggw_kernel(const float* __restrict__ att, const float* __restrict__ weight,
                            __bf16* __restrict__ hi, __bf16* __restrict__ lo) {
  int ci = threadIdx.x;
  int co = blockIdx.x;
  int t  = blockIdx.y;
  int b  = blockIdx.z;
  size_t wbase = ((size_t)co * CIN_ + ci) * NTAP + t;
  float v = 0.f;
  for (int k = 0; k < KEXP_; ++k)
    v += att[b * KEXP_ + k] * weight[(size_t)k * COUT_ * CIN_ * NTAP + wbase];
  __bf16 h = (__bf16)v;
  __bf16 l = (__bf16)(v - (float)h);
  size_t o = (((size_t)(b * NTAP + t) * COUT_) + co) * CIN_ + ci;
  hi[o] = h;
  lo[o] = l;
}

// ---------------- Phase 3: implicit-GEMM conv via bf16 WMMA (hi/lo split) ----------------
// block = (oh, m_blk, b); 256 threads = 8 waves; each wave: 2x2 tiles of 16x16.
__global__ __launch_bounds__(256)
void conv_kernel(const float* __restrict__ x, const __bf16* __restrict__ wAhi,
                 const __bf16* __restrict__ wAlo, const float* __restrict__ aggB,
                 float* __restrict__ out) {
  constexpr int APITCH = 40;  // padded pitch (bf16 elems), 80B: 16B-aligned chunks, spreads banks
  constexpr int XPITCH = 40;
  __shared__ __bf16 ah[128 * APITCH];
  __shared__ __bf16 al[128 * APITCH];
  __shared__ __bf16 xh[3 * 64 * XPITCH];
  __shared__ __bf16 xl[3 * 64 * XPITCH];

  const int oh   = blockIdx.x;        // one output row per block (N tile = 64)
  const int mblk = blockIdx.y;        // 128 Cout rows per block
  const int b    = blockIdx.z;
  const int tid  = threadIdx.x;
  const int lane = tid & 31;
  const int wave = tid >> 5;
  const int wm   = wave & 3;          // 4 waves over M (32 rows each)
  const int wn   = wave >> 2;         // 2 waves over N (32 cols each)
  const int lrow = lane & 15;
  const int lhalf = lane >> 4;

  v8f acc[2][2];
  const v8f vzero = {0.f, 0.f, 0.f, 0.f, 0.f, 0.f, 0.f, 0.f};
#pragma unroll
  for (int mi = 0; mi < 2; ++mi)
#pragma unroll
    for (int ni = 0; ni < 2; ++ni) acc[mi][ni] = vzero;

  for (int cc = 0; cc < 8; ++cc) {    // 8 ci-chunks of 32 channels
    const int ci0 = cc * 32;
    __syncthreads();                  // previous iteration done reading x tiles
    // ---- stage x rows (oh-1..oh+1), transposed [row][col][ci], split hi/lo ----
    for (int idx = tid; idx < 3 * 32 * 64; idx += 256) {
      int col  = idx & 63;
      int rest = idx >> 6;
      int ci   = rest & 31;
      int r    = rest >> 5;
      int ih   = oh + r - 1;
      float v  = 0.f;
      if ((unsigned)ih < (unsigned)HW_) {
        size_t gidx = (((size_t)(b * CIN_ + ci0 + ci)) * HW_ + ih) * HW_ + col;
        v = x[gidx];
        // prefetch next ci-chunk's row into cache (global_prefetch_b8)
        if (cc < 7 && r == 1)
          __builtin_prefetch(&x[gidx + (size_t)32 * HW_ * HW_], 0, 1);
      }
      __bf16 h = (__bf16)v;
      __bf16 l = (__bf16)(v - (float)h);
      int o = (r * 64 + col) * XPITCH + ci;
      xh[o] = h;
      xl[o] = l;
    }
    for (int t = 0; t < NTAP; ++t) {
      __syncthreads();                // x staged / previous A done being read
      // ---- stage A tile [128 co][32 ci] for this tap (bf16 already) ----
      {
        const __bf16* gh = wAhi + (((size_t)(b * NTAP + t) * COUT_) + mblk * 128) * CIN_ + ci0;
        const __bf16* gl = wAlo + (((size_t)(b * NTAP + t) * COUT_) + mblk * 128) * CIN_ + ci0;
#if USE_ASYNC_LDS
        for (int idx = tid; idx < 512; idx += 256) {   // 512 chunks of 16B each (hi & lo)
          int row = idx >> 2, ch = idx & 3;
          __builtin_amdgcn_global_load_async_to_lds_b128(
              (GlobalV4*)(gh + (size_t)row * CIN_ + ch * 8),
              (LdsV4*)&ah[row * APITCH + ch * 8], 0, 0);
          __builtin_amdgcn_global_load_async_to_lds_b128(
              (GlobalV4*)(gl + (size_t)row * CIN_ + ch * 8),
              (LdsV4*)&al[row * APITCH + ch * 8], 0, 0);
        }
        __builtin_amdgcn_s_wait_asynccnt(0);
#else
        for (int idx = tid; idx < 512; idx += 256) {   // 512 chunks of 16B each (hi & lo)
          int row = idx >> 2, ch = idx & 3;
          *(u32x4*)&ah[row * APITCH + ch * 8] = *(const u32x4*)(gh + (size_t)row * CIN_ + ch * 8);
          *(u32x4*)&al[row * APITCH + ch * 8] = *(const u32x4*)(gl + (size_t)row * CIN_ + ch * 8);
        }
#endif
      }
      __syncthreads();
      const int kh = t / 3, kw = t % 3;

      // ---- build fragments ----
      FragAB fah[2], fal[2], fbh[2], fbl[2];
#pragma unroll
      for (int mi = 0; mi < 2; ++mi) {
        int arow = wm * 32 + mi * 16 + lrow;         // A: lane holds row M=lane%16
        int kb   = lhalf ? 8 : 0;                    // K 0-7/16-23 vs 8-15/24-31
        int base = arow * APITCH + kb;
        fah[mi].u[0] = *(const u32x4*)&ah[base];
        fah[mi].u[1] = *(const u32x4*)&ah[base + 16];
        fal[mi].u[0] = *(const u32x4*)&al[base];
        fal[mi].u[1] = *(const u32x4*)&al[base + 16];
      }
      const u32x4 z4 = {0u, 0u, 0u, 0u};
#pragma unroll
      for (int ni = 0; ni < 2; ++ni) {
        int ow   = wn * 32 + ni * 16 + lrow;         // B: lane holds column N=lane%16
        int gcol = ow + kw - 1;                      // shifted input column (pad=1)
        bool valid = (unsigned)gcol < (unsigned)HW_;
        int colc = valid ? gcol : 0;
        int kk   = lhalf ? 8 : 0;                    // K 0-7/16-23 vs 8-15/24-31
        int base = (kh * 64 + colc) * XPITCH + kk;
        u32x4 q0 = *(const u32x4*)&xh[base];
        u32x4 q1 = *(const u32x4*)&xh[base + 16];
        u32x4 p0 = *(const u32x4*)&xl[base];
        u32x4 p1 = *(const u32x4*)&xl[base + 16];
        fbh[ni].u[0] = valid ? q0 : z4;              // per-lane select keeps EXEC all-1
        fbh[ni].u[1] = valid ? q1 : z4;
        fbl[ni].u[0] = valid ? p0 : z4;
        fbl[ni].u[1] = valid ? p1 : z4;
      }
      // ---- 3-product bf16 split: AhBh + AhBl + AlBh, f32 accumulate ----
#pragma unroll
      for (int mi = 0; mi < 2; ++mi)
#pragma unroll
        for (int ni = 0; ni < 2; ++ni) {
          acc[mi][ni] = __builtin_amdgcn_wmma_f32_16x16x32_bf16(
              false, fah[mi].v, false, fbh[ni].v, (short)0, acc[mi][ni], false, false);
          acc[mi][ni] = __builtin_amdgcn_wmma_f32_16x16x32_bf16(
              false, fah[mi].v, false, fbl[ni].v, (short)0, acc[mi][ni], false, false);
          acc[mi][ni] = __builtin_amdgcn_wmma_f32_16x16x32_bf16(
              false, fal[mi].v, false, fbh[ni].v, (short)0, acc[mi][ni], false, false);
        }
    }
  }

  // ---- store C + bias; C layout: lanes<16: M=vgpr, lanes>=16: M=vgpr+8; N=lane%16 ----
#pragma unroll
  for (int mi = 0; mi < 2; ++mi)
#pragma unroll
    for (int ni = 0; ni < 2; ++ni) {
      int ow = wn * 32 + ni * 16 + lrow;
#pragma unroll
      for (int v = 0; v < 8; ++v) {
        int co = mblk * 128 + wm * 32 + mi * 16 + v + (lhalf ? 8 : 0);
        float bia = aggB[b * COUT_ + co];
        out[(((size_t)(b * COUT_ + co)) * HW_ + oh) * HW_ + ow] = acc[mi][ni][v] + bia;
      }
    }
}

extern "C" void kernel_launch(void* const* d_in, const int* in_sizes, int n_in,
                              void* d_out, int out_size, void* d_ws, size_t ws_size,
                              hipStream_t stream) {
  const float* x     = (const float*)d_in[0];
  const float* fc1w  = (const float*)d_in[1];
  const float* fc1b  = (const float*)d_in[2];
  const float* fc2w  = (const float*)d_in[3];
  const float* fc2b  = (const float*)d_in[4];
  const float* weight= (const float*)d_in[5];
  const float* bias  = (const float*)d_in[6];
  float* out = (float*)d_out;

  char* ws = (char*)d_ws;
  float* att  = (float*)ws;                 // 32*4 floats
  float* aggB = (float*)(ws + 512);         // 32*256 floats
  const size_t AELEMS = (size_t)B_ * NTAP * COUT_ * CIN_;  // 18,874,368 bf16 per array
  __bf16* wAhi = (__bf16*)(ws + 65536);
  __bf16* wAlo = wAhi + AELEMS;

  attn_kernel<<<B_, 256, 0, stream>>>(x, fc1w, fc1b, fc2w, fc2b, att);
  aggb_kernel<<<B_, COUT_, 0, stream>>>(att, bias, aggB);
  aggw_kernel<<<dim3(COUT_, NTAP, B_), CIN_, 0, stream>>>(att, weight, wAhi, wAlo);
  conv_kernel<<<dim3(HW_, 2, B_), 256, 0, stream>>>(x, wAhi, wAlo, aggB, out);
}
